// qLayer_42107859370002
// MI455X (gfx1250) — matrix operand order repaired and checked by reading, not compiled, not run on hardware
//
#include <hip/hip_runtime.h>

typedef __attribute__((ext_vector_type(16))) _Float16 v16h;
typedef __attribute__((ext_vector_type(8)))  _Float16 v8h;
typedef __attribute__((ext_vector_type(4)))  _Float16 v4h;
typedef __attribute__((ext_vector_type(8)))  float    v8f;

#define K_DIM 4096
#define N_DIM 11008
#define B_DIM 256

// ---------------------------------------------------------------------------
// Kernel 1: rowsum[b] = sum_k x[b,k]   (256 blocks x 256 threads)
// ---------------------------------------------------------------------------
__global__ __launch_bounds__(256) void rowsum_kernel(const float* __restrict__ x,
                                                     float* __restrict__ s) {
    __shared__ float red[256];
    const int b = blockIdx.x;
    const int t = threadIdx.x;
    float acc = 0.0f;
#pragma unroll
    for (int i = 0; i < K_DIM / 256; ++i)
        acc += x[(size_t)b * K_DIM + t + i * 256];
    red[t] = acc;
    __syncthreads();
#pragma unroll
    for (int off = 128; off > 0; off >>= 1) {
        if (t < off) red[t] += red[t + off];
        __syncthreads();
    }
    if (t == 0) s[b] = red[0];
}

// ---------------------------------------------------------------------------
// Kernel 2: double-buffered WMMA GEMM on int4 nibbles (exact in f16).
// Block tile 128M x 128N, 8 waves, each wave 64M x 32N (4x2 16x16 tiles).
// Software pipeline: global loads for step i+1 staged in VGPRs while
// WMMAs consume LDS buffer i.
// ---------------------------------------------------------------------------
#define LDA 40   // padded K-stride (halfs): 80B rows, 16B aligned
#define LDB 40

__global__ __launch_bounds__(256) void
q4gemm_wmma_kernel(const float* __restrict__ x,
                   const int*   __restrict__ qw,
                   const float* __restrict__ scales,
                   const float* __restrict__ zeroes,
                   const float* __restrict__ rowsum,
                   float*       __restrict__ out) {
    __shared__ _Float16 As[2][128][LDA];   // [buf][m][k]
    __shared__ _Float16 Bs[2][128][LDB];   // [buf][n][k]
    __shared__ float    Ssum[128];

    const int tid    = threadIdx.x;
    const int lane   = tid & 31;
    const int wid    = tid >> 5;        // 0..7
    const int wave_m = wid & 1;         // 2 waves along M (64 rows each)
    const int wave_n = wid >> 1;        // 4 waves along N (32 cols each)

    const int m0 = blockIdx.y * 128;
    const int n0 = blockIdx.x * 128;

    if (tid < 128) Ssum[tid] = rowsum[m0 + tid];

    // ---- per-thread staging geometry (constant across K loop) ----
    // x tile: 128 rows x 8 float4-cols; thread covers rows tid>>3 + i*32
    const int xrow = tid >> 3;          // 0..31 (first of 4 rows, +32 apart)
    const int xc4  = tid & 7;           // float4 column
    const float* xp = x + (size_t)(m0 + xrow) * K_DIM + xc4 * 4;

    // q tile: 8 packed-row-pairs x 32 column-quads; one unit per thread
    const int qj = tid >> 5;            // row-pair 0..7 -> packed rows 2j,2j+1
    const int qc = (tid & 31) * 4;      // column quad base 0..124
    const int* qp = qw + (size_t)(2 * qj) * N_DIM + n0 + qc;

    v8f acc[4][2];
#pragma unroll
    for (int mt = 0; mt < 4; ++mt)
#pragma unroll
        for (int nt = 0; nt < 2; ++nt)
            acc[mt][nt] = (v8f){0.f, 0.f, 0.f, 0.f, 0.f, 0.f, 0.f, 0.f};

    const int c0 = (lane < 16) ? 0 : 8;    // A frag K-base (ISA A layout)
    const int kh = (lane < 16) ? 0 : 16;   // unused: B frags use c0, same K order as A
    const int ml = lane & 15;

    // ---- pipeline prologue: stage tile 0 into registers ----
    float4 xs[4];
    int4   qs[2];
#pragma unroll
    for (int i = 0; i < 4; ++i)
        xs[i] = *reinterpret_cast<const float4*>(xp + (size_t)i * 32 * K_DIM);
    qs[0] = *reinterpret_cast<const int4*>(qp);
    qs[1] = *reinterpret_cast<const int4*>(qp + N_DIM);
    xp += 32;                              // next K chunk
    qp += (size_t)16 * N_DIM;              // next 16 packed rows

    for (int kk = 0; kk < K_DIM; kk += 32) {
        const int buf = (kk >> 5) & 1;

        // ---- drain staged registers into LDS buffer `buf` ----
#pragma unroll
        for (int i = 0; i < 4; ++i) {
            v4h h = { (_Float16)xs[i].x, (_Float16)xs[i].y,
                      (_Float16)xs[i].z, (_Float16)xs[i].w };
            *reinterpret_cast<v4h*>(&As[buf][xrow + i * 32][xc4 * 4]) = h;
        }
#pragma unroll
        for (int c = 0; c < 4; ++c) {
            const unsigned a = ((const unsigned*)&qs[0])[c];
            const unsigned b = ((const unsigned*)&qs[1])[c];
            v4h h = { (_Float16)(int)(a & 15u), (_Float16)(int)((a >> 4) & 15u),
                      (_Float16)(int)(b & 15u), (_Float16)(int)((b >> 4) & 15u) };
            *reinterpret_cast<v4h*>(&Bs[buf][qc + c][4 * qj]) = h;
        }

        __syncthreads();

        // ---- stage next tile's global loads (overlaps with WMMAs below) ----
        if (kk + 32 < K_DIM) {
#pragma unroll
            for (int i = 0; i < 4; ++i)
                xs[i] = *reinterpret_cast<const float4*>(xp + (size_t)i * 32 * K_DIM);
            qs[0] = *reinterpret_cast<const int4*>(qp);
            qs[1] = *reinterpret_cast<const int4*>(qp + N_DIM);
            xp += 32;
            qp += (size_t)16 * N_DIM;
        }

        // ---- fragments + 8 WMMAs from LDS buffer `buf` ----
        v16h afrag[4];
#pragma unroll
        for (int mt = 0; mt < 4; ++mt) {
            const int m = wave_m * 64 + mt * 16 + ml;
            v8h alo = *reinterpret_cast<const v8h*>(&As[buf][m][c0]);
            v8h ahi = *reinterpret_cast<const v8h*>(&As[buf][m][c0 + 16]);
            afrag[mt] = __builtin_shufflevector(alo, ahi,
                0, 1, 2, 3, 4, 5, 6, 7, 8, 9, 10, 11, 12, 13, 14, 15);
        }
#pragma unroll
        for (int nt = 0; nt < 2; ++nt) {
            const int n = wave_n * 32 + nt * 16 + ml;
            v8h blo = *reinterpret_cast<const v8h*>(&Bs[buf][n][c0]);
            v8h bhi = *reinterpret_cast<const v8h*>(&Bs[buf][n][c0 + 16]);
            v16h bfrag = __builtin_shufflevector(blo, bhi,
                0, 1, 2, 3, 4, 5, 6, 7, 8, 9, 10, 11, 12, 13, 14, 15);
#pragma unroll
            for (int mt = 0; mt < 4; ++mt) {
                acc[mt][nt] = __builtin_amdgcn_wmma_f32_16x16x32_f16(
                    false, afrag[mt], false, bfrag,
                    (short)0, acc[mt][nt], false, false);
            }
        }
    }

    // ---- epilogue: out = acc*scale - rowsum*zero ----
#pragma unroll
    for (int nt = 0; nt < 2; ++nt) {
        const int ng = n0 + wave_n * 32 + nt * 16 + ml;
        const float sc = scales[ng];
        const float zr = zeroes[ng];
#pragma unroll
        for (int mt = 0; mt < 4; ++mt) {
            const int mbase = wave_m * 64 + mt * 16 + ((lane < 16) ? 0 : 8);
#pragma unroll
            for (int v = 0; v < 8; ++v) {
                const int mg = m0 + mbase + v;
                out[(size_t)mg * N_DIM + ng] =
                    acc[mt][nt][v] * sc - Ssum[mbase + v] * zr;
            }
        }
    }
}

// ---------------------------------------------------------------------------
extern "C" void kernel_launch(void* const* d_in, const int* in_sizes, int n_in,
                              void* d_out, int out_size, void* d_ws, size_t ws_size,
                              hipStream_t stream) {
    const float* x      = (const float*)d_in[0];
    const int*   qw     = (const int*)  d_in[1];
    const float* scales = (const float*)d_in[2];
    const float* zeroes = (const float*)d_in[3];
    float*       out    = (float*)d_out;
    float*       rowsum = (float*)d_ws;   // 256 floats of scratch

    rowsum_kernel<<<B_DIM, 256, 0, stream>>>(x, rowsum);

    dim3 grid(N_DIM / 128, B_DIM / 128);   // (86, 2)
    q4gemm_wmma_kernel<<<grid, 256, 0, stream>>>(x, qw, scales, zeroes, rowsum, out);
}
